// MultiHeadedAttention_54271206752303
// MI455X (gfx1250) — matrix operand order repaired and checked by reading, not compiled, not run on hardware
//
#include <hip/hip_runtime.h>
#include <hip/hip_bf16.h>

typedef __attribute__((ext_vector_type(16))) __bf16 v16bf;
typedef __attribute__((ext_vector_type(8)))  __bf16 v8bf;
typedef __attribute__((ext_vector_type(8)))  float  v8f;

static constexpr int kB      = 2;
static constexpr int kS      = 2048;
static constexpr int kD      = 1024;
static constexpr int kH      = 16;
static constexpr int kDK     = 64;

// native conversions: lower to v_cvt_pk_bf16_f32 on gfx1250
__device__ __forceinline__ __bf16 f2bf(float f) { return (__bf16)f; }
__device__ __forceinline__ float  bf2f(__bf16 b) { return (float)b; }

__device__ __forceinline__ v8f wmma_bf16(v16bf a, v16bf b, v8f c) {
  // D = A(16x32 bf16) * B(32x16 bf16) + C(16x16 f32)
  return __builtin_amdgcn_wmma_f32_16x16x32_bf16(
      false, a, false, b, (short)0, c, false, false);
}

__device__ __forceinline__ v16bf combine16(v8bf lo, v8bf hi) {
  return __builtin_shufflevector(lo, hi, 0, 1, 2, 3, 4, 5, 6, 7,
                                 8, 9, 10, 11, 12, 13, 14, 15);
}

// two contiguous 8-element bf16 runs (16B-aligned each)
__device__ __forceinline__ v16bf ld_bf16_2x8(const __bf16* p0, const __bf16* p1) {
  v8bf lo = *(const v8bf*)p0;
  v8bf hi = *(const v8bf*)p1;
  return combine16(lo, hi);
}

// 8 contiguous f32 -> 8 bf16
__device__ __forceinline__ v8bf cvt8(const float* p) {
  float4 a = *(const float4*)p;
  float4 b = *(const float4*)(p + 4);
  v8bf r;
  r[0] = f2bf(a.x); r[1] = f2bf(a.y); r[2] = f2bf(a.z); r[3] = f2bf(a.w);
  r[4] = f2bf(b.x); r[5] = f2bf(b.y); r[6] = f2bf(b.z); r[7] = f2bf(b.w);
  return r;
}

// ---------------- elementwise f32 -> bf16 ----------------
__global__ void cvt_f32_bf16_kernel(const float* __restrict__ in,
                                    __bf16* __restrict__ out, int n) {
  int i = blockIdx.x * blockDim.x + threadIdx.x;
  int stride = gridDim.x * blockDim.x;
  for (; i < n; i += stride) out[i] = f2bf(in[i]);
}

// ---------------- GEMM: out = A(MxK) @ W^T, W is (NxK) row-major bf16 -------
// one wave computes a 32x64 output block: 2 M-tiles x 4 N-tiles
// MODE 0: bf16 out scattered to [b,h,s,dk]
// MODE 1: bf16 out scattered to [b,h,dk,s]  (transposed V for attention)
// MODE 2: f32 out row-major [M,N]
template <bool AF32, int MODE>
__global__ __launch_bounds__(32) void gemm_wmma_kernel(
    const void* __restrict__ Aptr, const __bf16* __restrict__ Bw,
    void* __restrict__ out, int M, int N, int K) {
  const int lane = threadIdx.x;
  const int nb   = lane & 15;
  const int sub  = lane >> 4;
  const int mtile0 = blockIdx.y * 2;
  const int ntile0 = blockIdx.x * 4;

  v8f acc[2][4];
#pragma unroll
  for (int mt = 0; mt < 2; ++mt)
#pragma unroll
    for (int t = 0; t < 4; ++t) acc[mt][t] = v8f{};

  for (int kc = 0; kc < K; kc += 32) {
    v16bf a[2];
    v16bf bv[4];
    // hoist all loads: one long clause, progressive waits
#pragma unroll
    for (int t = 0; t < 4; ++t) {
      const int gn = (ntile0 + t) * 16 + nb;     // output column == W row
      const __bf16* bp = Bw + (size_t)gn * K + kc + sub * 8;
      bv[t] = ld_bf16_2x8(bp, bp + 16);
    }
#pragma unroll
    for (int mt = 0; mt < 2; ++mt) {
      const int gm = (mtile0 + mt) * 16 + nb;   // A row (A-layout)
      const int o0 = gm * K + kc + sub * 8;
      const int o1 = gm * K + kc + 16 + sub * 8;
      if constexpr (AF32) {
        const float* A = (const float*)Aptr;
        a[mt] = combine16(cvt8(A + o0), cvt8(A + o1));
      } else {
        const __bf16* A = (const __bf16*)Aptr;
        a[mt] = ld_bf16_2x8(A + o0, A + o1);
      }
    }
#pragma unroll
    for (int t = 0; t < 4; ++t)
#pragma unroll
      for (int mt = 0; mt < 2; ++mt)
        acc[mt][t] = wmma_bf16(a[mt], bv[t], acc[mt][t]);
  }

#pragma unroll
  for (int mt = 0; mt < 2; ++mt) {
#pragma unroll
    for (int t = 0; t < 4; ++t) {
#pragma unroll
      for (int r = 0; r < 8; ++r) {
        const int m = (mtile0 + mt) * 16 + r + sub * 8;
        const int n = (ntile0 + t) * 16 + nb;
        const float v = acc[mt][t][r];
        if constexpr (MODE == 2) {
          ((float*)out)[(size_t)m * N + n] = v;
        } else {
          const int bb = m / kS, s = m % kS;
          const int h = n / kDK, dk = n % kDK;
          __bf16* o = (__bf16*)out;
          if constexpr (MODE == 0)
            o[(((size_t)(bb * kH + h)) * kS + s) * kDK + dk] = f2bf(v);
          else
            o[(((size_t)(bb * kH + h)) * kDK + dk) * kS + s] = f2bf(v);
        }
      }
    }
  }
}

// ---------------- RoPE in place on [b,h,s,dk] bf16 ----------------
__global__ void rope_kernel(__bf16* __restrict__ qh,
                            const float* __restrict__ cosp,
                            const float* __restrict__ sinp, int total) {
  int idx = blockIdx.x * blockDim.x + threadIdx.x;
  if (idx >= total) return;                 // total = B*H*S*(DK/2)
  const int i  = idx & 31;
  const int s  = (idx >> 5) & (kS - 1);
  const int bh = idx >> 16;                 // 5 + log2(2048)
  const size_t base = ((size_t)bh * kS + s) * kDK;
  const float c   = cosp[s * kDK + i];
  const float sn  = sinp[s * kDK + i];
  const float c2  = cosp[s * kDK + i + 32];
  const float sn2 = sinp[s * kDK + i + 32];
  const float x1 = bf2f(qh[base + i]);
  const float x2 = bf2f(qh[base + i + 32]);
  qh[base + i]      = f2bf(x1 * c - x2 * sn);
  qh[base + i + 32] = f2bf(x2 * c2 + x1 * sn2);
}

// ---------------- flash attention: one wave per (b,h,16-query tile) --------
__global__ __launch_bounds__(32) void fa_kernel(
    const __bf16* __restrict__ qh, const __bf16* __restrict__ kh,
    const __bf16* __restrict__ vt, __bf16* __restrict__ ctx) {
  __shared__ float sS[16][36];   // pad 36 -> conflict-free row access
  __shared__ float sPmax[2][16];
  __shared__ float sPsum[2][16];
  __shared__ float sF[16];
  __shared__ float sInv[16];

  const int lane = threadIdx.x;
  const int nb   = lane & 15;
  const int sub  = lane >> 4;
  const int qtile = blockIdx.x;
  const int bh    = blockIdx.y;
  const int bb = bh / kH, h = bh % kH;
  const int qg0 = qtile * 16;

  // Q tile in A-layout (DK=64 -> two K=32 chunks)
  const size_t qrow = ((size_t)bh * kS + qg0 + nb) * kDK;
  v16bf qa0 = ld_bf16_2x8(qh + qrow + 0 + sub * 8, qh + qrow + 16 + sub * 8);
  v16bf qa1 = ld_bf16_2x8(qh + qrow + 32 + sub * 8, qh + qrow + 48 + sub * 8);

  v8f acc[4];
#pragma unroll
  for (int t = 0; t < 4; ++t) acc[t] = v8f{};

  float rowmax = -1e30f, rowsum = 0.f;
  const int row = nb;
  const int kend = qg0 + 16;
  const float scale = 0.125f;     // 1/sqrt(64)

  for (int kb = 0; kb < kend; kb += 32) {
    // ---- hoisted K-tile loads (one clause, progressive waits) ----
    v16bf kv[2][2];
#pragma unroll
    for (int t = 0; t < 2; ++t) {
      const size_t krow = ((size_t)bh * kS + kb + t * 16 + nb) * kDK;
      const __bf16* kp0 = kh + krow + sub * 8;
      const __bf16* kp1 = kh + krow + 32 + sub * 8;
      kv[t][0] = ld_bf16_2x8(kp0, kp0 + 16);
      kv[t][1] = ld_bf16_2x8(kp1, kp1 + 16);
    }
    v8f sc[2];
#pragma unroll
    for (int t = 0; t < 2; ++t) {
      sc[t] = wmma_bf16(qa0, kv[t][0], v8f{});
      sc[t] = wmma_bf16(qa1, kv[t][1], sc[t]);
    }
    // ---- V-tile loads issued early: latency hidden under softmax ----
    v16bf vb[4];
#pragma unroll
    for (int t = 0; t < 4; ++t) {
      const int dk = t * 16 + nb;
      const __bf16* vp = vt + ((size_t)bh * kDK + dk) * kS + kb + sub * 8;
      vb[t] = ld_bf16_2x8(vp, vp + 16);
    }
    // ---- scale + causal mask, D-layout -> LDS [m][k_local] ----
#pragma unroll
    for (int t = 0; t < 2; ++t) {
#pragma unroll
      for (int r = 0; r < 8; ++r) {
        const int m  = r + sub * 8;
        const int kg = kb + t * 16 + nb;
        float v = sc[t][r] * scale;
        if (kg > qg0 + m) v = -1e9f;
        sS[m][t * 16 + nb] = v;
      }
    }
    __syncthreads();
    // ---- online softmax: each half reduces its own 16 columns ----
    float pmax = -1e30f;
#pragma unroll
    for (int j = 0; j < 16; ++j) pmax = fmaxf(pmax, sS[row][sub * 16 + j]);
    sPmax[sub][row] = pmax;
    __syncthreads();
    const float rm_new = fmaxf(rowmax, fmaxf(sPmax[0][row], sPmax[1][row]));
    const float f = __expf(rowmax - rm_new);
    if (lane < 16) sF[row] = f;
    float psum = 0.f;
#pragma unroll
    for (int j = 0; j < 16; ++j) {
      const float p = __expf(sS[row][sub * 16 + j] - rm_new);
      sS[row][sub * 16 + j] = p;
      psum += p;
    }
    sPsum[sub][row] = psum;
    __syncthreads();
    rowsum = rowsum * f + sPsum[0][row] + sPsum[1][row];
    rowmax = rm_new;
    // ---- rescale context accumulators ----
#pragma unroll
    for (int t = 0; t < 4; ++t) {
#pragma unroll
      for (int r = 0; r < 8; ++r) acc[t][r] *= sF[r + sub * 8];
    }
    // ---- P in A-layout bf16 from LDS ----
    v16bf pa;
    {
      const float* prow = &sS[nb][0];
#pragma unroll
      for (int j = 0; j < 8; ++j) {
        pa[j]     = f2bf(prow[sub * 8 + j]);
        pa[8 + j] = f2bf(prow[16 + sub * 8 + j]);
      }
    }
    // ---- ctx += P(16x32) @ V(32x64), V tiles already in registers ----
#pragma unroll
    for (int t = 0; t < 4; ++t) acc[t] = wmma_bf16(pa, vb[t], acc[t]);
    __syncthreads();
  }

  if (lane < 16) sInv[row] = 1.f / rowsum;
  __syncthreads();
#pragma unroll
  for (int t = 0; t < 4; ++t) {
#pragma unroll
    for (int r = 0; r < 8; ++r) {
      const int m  = r + sub * 8;
      const int qg = qg0 + m;
      const int dk = t * 16 + nb;
      ctx[((size_t)(bb * kS + qg)) * kD + h * kDK + dk] =
          f2bf(acc[t][r] * sInv[m]);
    }
  }
}

extern "C" void kernel_launch(void* const* d_in, const int* in_sizes, int n_in,
                              void* d_out, int out_size, void* d_ws, size_t ws_size,
                              hipStream_t stream) {
  (void)in_sizes; (void)n_in; (void)out_size; (void)ws_size;
  const float* x_q  = (const float*)d_in[0];
  const float* x_k  = (const float*)d_in[1];
  const float* x_v  = (const float*)d_in[2];
  const float* Wq   = (const float*)d_in[3];
  const float* Wk   = (const float*)d_in[4];
  const float* Wv   = (const float*)d_in[5];
  const float* Wo   = (const float*)d_in[6];
  const float* cosp = (const float*)d_in[7];
  const float* sinp = (const float*)d_in[8];
  // d_in[9] (mask) unused: causality is static

  __bf16* ws = (__bf16*)d_ws;
  const size_t WELEM = (size_t)kD * kD;                 // 1 Mi
  const size_t QKV   = (size_t)kB * kH * kS * kDK;      // 4 Mi
  __bf16* Wq_b = ws;
  __bf16* Wk_b = Wq_b + WELEM;
  __bf16* Wv_b = Wk_b + WELEM;
  __bf16* Wo_b = Wv_b + WELEM;
  __bf16* q_h  = Wo_b + WELEM;   // [b,h,s,dk]
  __bf16* k_h  = q_h + QKV;      // [b,h,s,dk]
  __bf16* v_t  = k_h + QKV;      // [b,h,dk,s]
  __bf16* c_ws = v_t + QKV;      // [b,s,d]

  // 1) weights -> bf16
  cvt_f32_bf16_kernel<<<1024, 256, 0, stream>>>(Wq, Wq_b, (int)WELEM);
  cvt_f32_bf16_kernel<<<1024, 256, 0, stream>>>(Wk, Wk_b, (int)WELEM);
  cvt_f32_bf16_kernel<<<1024, 256, 0, stream>>>(Wv, Wv_b, (int)WELEM);
  cvt_f32_bf16_kernel<<<1024, 256, 0, stream>>>(Wo, Wo_b, (int)WELEM);

  // 2) projections: one wave per 32x64 output block
  const int M = kB * kS;                  // 4096
  dim3 gblk(32, 1, 1);
  dim3 ggrid(kD / 64, M / 32, 1);         // (16, 128)
  gemm_wmma_kernel<true, 0><<<ggrid, gblk, 0, stream>>>(x_q, Wq_b, q_h, M, kD, kD);
  gemm_wmma_kernel<true, 0><<<ggrid, gblk, 0, stream>>>(x_k, Wk_b, k_h, M, kD, kD);
  gemm_wmma_kernel<true, 1><<<ggrid, gblk, 0, stream>>>(x_v, Wv_b, v_t, M, kD, kD);

  // 3) RoPE on Q and K
  const int rtotal = kB * kH * kS * (kDK / 2);
  rope_kernel<<<(rtotal + 255) / 256, 256, 0, stream>>>(q_h, cosp, sinp, rtotal);
  rope_kernel<<<(rtotal + 255) / 256, 256, 0, stream>>>(k_h, cosp, sinp, rtotal);

  // 4) flash attention
  dim3 fgrid(kS / 16, kB * kH, 1);
  fa_kernel<<<fgrid, gblk, 0, stream>>>(q_h, k_h, v_t, c_ws);

  // 5) output projection (f32 out)
  gemm_wmma_kernel<false, 2><<<ggrid, gblk, 0, stream>>>(c_ws, Wo_b, d_out, M, kD, kD);
}
